// EvidenceConstrainedSelfAttention_82961588289563
// MI455X (gfx1250) — matrix operand order repaired and checked by reading, not compiled, not run on hardware
//
#include <hip/hip_runtime.h>
#include <hip/hip_bf16.h>

// ---------------------------------------------------------------------------
// EvidenceConstrainedSelfAttention for MI455X (gfx1250, wave32, WMMA f16 +
// Tensor Data Mover staging).  B=4, S=2048, D=1024, H=16, DK=64
// ---------------------------------------------------------------------------

typedef __attribute__((ext_vector_type(16))) _Float16 v16h;
typedef __attribute__((ext_vector_type(8)))  float    v8f;
typedef __attribute__((ext_vector_type(4)))  unsigned int v4u;
typedef __attribute__((ext_vector_type(4)))  int      v4i;
typedef __attribute__((ext_vector_type(8)))  int      v8i;

#if __has_builtin(__builtin_amdgcn_tensor_load_to_lds)
#define ECSA_HAVE_TDM 1
#if __has_include(<hip/amd_detail/amd_gfx1250_TDM.h>)
#define ECSA_TDM_6ARG 1
#endif
#endif

namespace ecsa {

constexpr int Bb = 4, Ss = 2048, Dd = 1024, Hh = 16, DKk = 64;

union Frag {
    v16h v;
    uint4 q[2];
};

// A-matrix 16x32 f16 fragment (ISA 7.12.2): lane L holds row M=L%16;
// element e <-> K = (e<8 ? half*8+e : 16+half*8+(e-8)), half = L/16.
__device__ __forceinline__ v16h load_fragA(const _Float16* base, int stride_h,
                                           int m, int half_id, int koff) {
    Frag f;
    const _Float16* row = base + (size_t)m * stride_h + koff + half_id * 8;
    f.q[0] = *(const uint4*)(row);
    f.q[1] = *(const uint4*)(row + 16);
    return f.v;
}

// B-matrix 32x16 f16 fragment: lane L holds col N=L%16; element e <-> K =
// (e<8 ? half*8+e : 16+half*8+(e-8)).  base points at B^T (row n = column n of B).
__device__ __forceinline__ v16h load_fragB(const _Float16* base, int stride_h,
                                           int n, int half_id, int koff) {
    Frag f;
    const _Float16* row = base + (size_t)n * stride_h + koff + half_id * 8;
    f.q[0] = *(const uint4*)(row);
    f.q[1] = *(const uint4*)(row + 16);
    return f.v;
}

__device__ __forceinline__ v8f wmma16(v16h a, v16h b, v8f c) {
    return __builtin_amdgcn_wmma_f32_16x16x32_f16(false, a, false, b,
                                                  (short)0, c, false, false);
}

#if defined(ECSA_HAVE_TDM)
// 32-bit LDS byte offset of a __shared__ object (generic -> AS(3) -> int).
__device__ __forceinline__ unsigned lds_off(const void* p) {
    return (unsigned)(unsigned long long)
           (const __attribute__((address_space(3))) char*)p;
}

// Issue a 2-D TDM tile load (f16 elements) with LDS row padding.
//   tile: tile1 rows x tile0 elements; global row stride = stride0 elements
//   pad_interval/pad_amount: ISA D# codes (interval 2^(c+1) DW, amount a+1 DW)
__device__ __forceinline__ void tdm_load_2d(unsigned lds_addr, const void* gptr,
                                            unsigned tensor_d0, unsigned tensor_d1,
                                            unsigned tile0, unsigned tile1,
                                            unsigned stride0,
                                            unsigned pad_interval,
                                            unsigned pad_amount) {
    unsigned long long ga = (unsigned long long)(size_t)gptr;
    v4u g0;
    g0[0] = 1u;                                     // count=1, user descriptor
    g0[1] = lds_addr;                               // lds_addr[31:0]
    g0[2] = (unsigned)ga;                           // global_addr[31:0]
    g0[3] = (unsigned)((ga >> 32) & 0x01FFFFFFu)    // global_addr[56:32]
          | 0x80000000u;                            // type=2 ("image")
    v8i g1;
    g1[0] = (int)((1u << 16)                        // data_size = 2 bytes
                | (1u << 20)                        // pad_enable
                | (pad_interval << 22)
                | (pad_amount << 25));
    g1[1] = (int)((tensor_d0 & 0xFFFFu) << 16);     // atomic_addr=0 | d0.lo
    g1[2] = (int)((tensor_d0 >> 16) | ((tensor_d1 & 0xFFFFu) << 16));
    g1[3] = (int)((tensor_d1 >> 16) | (tile0 << 16));
    g1[4] = (int)(tile1 & 0xFFFFu);                 // tile_dim1 | tile_dim2=0
    g1[5] = (int)stride0;                           // dim0 stride lo32
    g1[6] = 0;
    g1[7] = 0;
    v4i z4 = {0, 0, 0, 0};
#if defined(ECSA_TDM_6ARG)
    v8i z8 = {0, 0, 0, 0, 0, 0, 0, 0};
    __builtin_amdgcn_tensor_load_to_lds(g0, g1, z4, z4, z8, 0);
#else
    __builtin_amdgcn_tensor_load_to_lds(g0, g1, z4, z4, 0);
#endif
}

__device__ __forceinline__ void wait_tensorcnt0() {
#if __has_builtin(__builtin_amdgcn_s_wait_tensorcnt)
    __builtin_amdgcn_s_wait_tensorcnt(0);
#else
    asm volatile("s_wait_tensorcnt 0x0" ::: "memory");
#endif
}
#endif // ECSA_HAVE_TDM

// ---------------------------------------------------------------------------
// Kernel 1: Y = X @ W^T  (X: [8192,1024] f32, W: [1024,1024] f32)
// 128x64 block tile; each of 8 waves owns a 16x64 strip (4 WMMA / K-step).
//   Q,K -> (B,H,S,DK): s_stride=DK, d_stride=1
//   V   -> (B,H,DK,S): s_stride=1,  d_stride=S   (pre-transposed for P@V)
// ---------------------------------------------------------------------------
__global__ __launch_bounds__(256)
void qkv_proj_kernel(const float* __restrict__ X, const float* __restrict__ W,
                     _Float16* __restrict__ Y, float out_scale,
                     int s_stride, int d_stride)
{
    __shared__ _Float16 Ash[128][40];
    __shared__ _Float16 Bsh[64][40];

    const int tid     = threadIdx.x;
    const int lane    = tid & 31;
    const int wave    = tid >> 5;
    const int half_id = lane >> 4;
    const int lrow    = lane & 15;

    const int n0 = blockIdx.x * 64;
    const int m0 = blockIdx.y * 128;

    v8f acc[4] = {};

    const int ar = tid >> 3, ac = (tid & 7) * 4;   // A staging coords
    const int br = tid >> 2, bc = (tid & 3) * 8;   // B staging coords

    for (int k0 = 0; k0 < Dd; k0 += 32) {
        __syncthreads();
#pragma unroll
        for (int i = 0; i < 4; ++i) {              // A: 128 x 32 (f32 -> f16)
            const float* xp = X + (size_t)(m0 + ar + 32 * i) * Dd + k0 + ac;
            float4 x = *(const float4*)xp;
            __builtin_prefetch(xp + 32, 0, 1);
            Ash[ar + 32 * i][ac + 0] = (_Float16)x.x;
            Ash[ar + 32 * i][ac + 1] = (_Float16)x.y;
            Ash[ar + 32 * i][ac + 2] = (_Float16)x.z;
            Ash[ar + 32 * i][ac + 3] = (_Float16)x.w;
        }
        {                                           // B: 64 x 32 (f32 -> f16)
            const float* wp = W + (size_t)(n0 + br) * Dd + k0 + bc;
            float4 w0 = *(const float4*)(wp);
            float4 w1 = *(const float4*)(wp + 4);
            __builtin_prefetch(wp + 32, 0, 1);
            Bsh[br][bc + 0] = (_Float16)w0.x;
            Bsh[br][bc + 1] = (_Float16)w0.y;
            Bsh[br][bc + 2] = (_Float16)w0.z;
            Bsh[br][bc + 3] = (_Float16)w0.w;
            Bsh[br][bc + 4] = (_Float16)w1.x;
            Bsh[br][bc + 5] = (_Float16)w1.y;
            Bsh[br][bc + 6] = (_Float16)w1.z;
            Bsh[br][bc + 7] = (_Float16)w1.w;
        }
        __syncthreads();

        v16h a = load_fragA(&Ash[wave * 16][0], 40, lrow, half_id, 0);
#pragma unroll
        for (int nt = 0; nt < 4; ++nt) {
            v16h b = load_fragB(&Bsh[nt * 16][0], 40, lrow, half_id, 0);
            acc[nt] = wmma16(a, b, acc[nt]);
        }
    }

#pragma unroll
    for (int nt = 0; nt < 4; ++nt) {
#pragma unroll
        for (int r = 0; r < 8; ++r) {
            int m = m0 + wave * 16 + r + 8 * half_id;
            int n = n0 + nt * 16 + lrow;
            int bb = m >> 11;            // / S
            int s  = m & (Ss - 1);
            int h  = n >> 6;             // / DK
            int dk = n & (DKk - 1);
            size_t off = (size_t)(bb * Hh + h) * (size_t)Ss * DKk
                       + (size_t)s * s_stride + (size_t)dk * d_stride;
            Y[off] = (_Float16)(acc[nt][r] * out_scale);
        }
    }
}

// ---------------------------------------------------------------------------
// Kernel 2: flash attention. grid = (B*H, S/128), block = 256 (8 waves).
// K / V^T chunks staged in LDS by the Tensor Data Mover (fallback: b128 copy).
// ---------------------------------------------------------------------------
__global__ __launch_bounds__(256)
void attn_kernel(const _Float16* __restrict__ Q, const _Float16* __restrict__ K,
                 const _Float16* __restrict__ VT, const float* __restrict__ mask,
                 _Float16* __restrict__ Ctx)
{
    __shared__ _Float16 Ksh[32][72];       // 32 keys x 64 dk   (+4DW row pad)
    __shared__ _Float16 VTsh[64][40];      // 64 dk x 32 keys   (+4DW row pad)
    __shared__ _Float16 Psh[8][16][40];    // per-wave P tile: 16 q x 32 keys

    const int tid     = threadIdx.x;
    const int lane    = tid & 31;
    const int wave    = tid >> 5;
    const int half_id = lane >> 4;
    const int lrow    = lane & 15;

    const int bh = blockIdx.x;             // b*H + h
    const int b  = bh >> 4;
    const int h  = bh & (Hh - 1);
    const int qs = blockIdx.y * 128 + wave * 16;

    // Q fragments (A layout), dk 0..31 / 32..63 — Q pre-scaled by 1/sqrt(DK)
    v16h qf0, qf1;
    {
        const _Float16* qrow = Q + ((size_t)bh * Ss + qs + lrow) * DKk;
        Frag f0, f1;
        f0.q[0] = *(const uint4*)(qrow + half_id * 8);
        f0.q[1] = *(const uint4*)(qrow + 16 + half_id * 8);
        f1.q[0] = *(const uint4*)(qrow + 32 + half_id * 8);
        f1.q[1] = *(const uint4*)(qrow + 48 + half_id * 8);
        qf0 = f0.v;
        qf1 = f1.v;
    }

    v8f o0 = {}, o1 = {}, o2 = {}, o3 = {};
    float m_r[8], l_r[8];
#pragma unroll
    for (int r = 0; r < 8; ++r) { m_r[r] = -1e30f; l_r[r] = 0.0f; }

    const _Float16* Kbase = K  + (size_t)bh * Ss * DKk;
    const _Float16* Vbase = VT + (size_t)bh * DKk * Ss;

#if defined(ECSA_HAVE_TDM)
    const unsigned kshOff  = lds_off(&Ksh[0][0]);
    const unsigned vtshOff = lds_off(&VTsh[0][0]);
#endif

    for (int j = 0; j < Ss; j += 32) {
        __syncthreads();
#if defined(ECSA_HAVE_TDM)
        if (wave == 0) {
            // K chunk: 32 rows x 64 f16, row stride 64; pad 32DW->+4DW
            tdm_load_2d(kshOff, Kbase + (size_t)j * DKk,
                        /*d0=*/DKk, /*d1=*/Ss, /*tile0=*/DKk, /*tile1=*/32,
                        /*stride0=*/DKk, /*pad_int=*/4, /*pad_amt=*/3);
            // V^T chunk: 64 rows x 32 f16, row stride S; pad 16DW->+4DW
            tdm_load_2d(vtshOff, Vbase + j,
                        /*d0=*/Ss, /*d1=*/DKk, /*tile0=*/32, /*tile1=*/DKk,
                        /*stride0=*/Ss, /*pad_int=*/3, /*pad_amt=*/3);
        }
        wait_tensorcnt0();
#else
        {   // cooperative staging fallback
            int r = tid >> 3;
            int c = (tid & 7) * 8;
            *(uint4*)&Ksh[r][c] =
                *(const uint4*)(Kbase + (size_t)(j + r) * DKk + c);
            int dr = tid >> 2;
            int dc = (tid & 3) * 8;
            *(uint4*)&VTsh[dr][dc] =
                *(const uint4*)(Vbase + (size_t)dr * Ss + j + dc);
        }
#endif
        __syncthreads();

        // logits: two 16-key tiles, DK=64 contraction (2 WMMA each)
        v8f s0 = {}, s1 = {};
        {
            v16h k00 = load_fragB(&Ksh[0][0],  72, lrow, half_id, 0);
            v16h k01 = load_fragB(&Ksh[0][0],  72, lrow, half_id, 32);
            s0 = wmma16(qf0, k00, s0);
            s0 = wmma16(qf1, k01, s0);
            v16h k10 = load_fragB(&Ksh[16][0], 72, lrow, half_id, 0);
            v16h k11 = load_fragB(&Ksh[16][0], 72, lrow, half_id, 32);
            s1 = wmma16(qf0, k10, s1);
            s1 = wmma16(qf1, k11, s1);
        }

        // mask + online softmax (row stats live in 16-lane halves)
        float p0[8], p1[8], alpha[8];
#pragma unroll
        for (int r = 0; r < 8; ++r) {
            const float* mp = mask + (size_t)(qs + r + 8 * half_id) * Ss + j;
            float l0 = s0[r] + mp[lrow];
            float l1 = s1[r] + mp[16 + lrow];
            float t = fmaxf(l0, l1);
            t = fmaxf(t, __shfl_xor(t, 1));
            t = fmaxf(t, __shfl_xor(t, 2));
            t = fmaxf(t, __shfl_xor(t, 4));
            t = fmaxf(t, __shfl_xor(t, 8));
            float newm = fmaxf(m_r[r], t);
            alpha[r] = __expf(m_r[r] - newm);
            m_r[r] = newm;
            p0[r] = __expf(l0 - newm);
            p1[r] = __expf(l1 - newm);
            float rs = p0[r] + p1[r];
            rs += __shfl_xor(rs, 1);
            rs += __shfl_xor(rs, 2);
            rs += __shfl_xor(rs, 4);
            rs += __shfl_xor(rs, 8);
            l_r[r] = l_r[r] * alpha[r] + rs;
        }

        // C-layout -> A-layout conversion for P via per-wave LDS
#pragma unroll
        for (int r = 0; r < 8; ++r) {
            int pr = r + 8 * half_id;
            Psh[wave][pr][lrow]      = (_Float16)p0[r];
            Psh[wave][pr][16 + lrow] = (_Float16)p1[r];
        }
#pragma unroll
        for (int r = 0; r < 8; ++r) {
            o0[r] *= alpha[r];
            o1[r] *= alpha[r];
            o2[r] *= alpha[r];
            o3[r] *= alpha[r];
        }

        // O += P @ V  (4 dk n-tiles)
        v16h pf  = load_fragA(&Psh[wave][0][0], 40, lrow, half_id, 0);
        v16h vf0 = load_fragB(&VTsh[0][0],  40, lrow, half_id, 0);
        v16h vf1 = load_fragB(&VTsh[16][0], 40, lrow, half_id, 0);
        v16h vf2 = load_fragB(&VTsh[32][0], 40, lrow, half_id, 0);
        v16h vf3 = load_fragB(&VTsh[48][0], 40, lrow, half_id, 0);
        o0 = wmma16(pf, vf0, o0);
        o1 = wmma16(pf, vf1, o1);
        o2 = wmma16(pf, vf2, o2);
        o3 = wmma16(pf, vf3, o3);
    }

    // epilogue: O / l, store ctx as (B, S, H*DK) f16
#pragma unroll
    for (int r = 0; r < 8; ++r) {
        float inv = 1.0f / l_r[r];
        int s = qs + r + 8 * half_id;
        _Float16* cp = Ctx + ((size_t)b * Ss + s) * (size_t)(Hh * DKk) + h * DKk;
        cp[ 0 + lrow] = (_Float16)(o0[r] * inv);
        cp[16 + lrow] = (_Float16)(o1[r] * inv);
        cp[32 + lrow] = (_Float16)(o2[r] * inv);
        cp[48 + lrow] = (_Float16)(o3[r] * inv);
    }
}

// ---------------------------------------------------------------------------
// Kernel 3: out = ctx @ Wo^T  (ctx f16 [8192,1024], Wo f32 [1024,1024])
// Same 128x64 block tile; f32 output straight into d_out.
// ---------------------------------------------------------------------------
__global__ __launch_bounds__(256)
void out_proj_kernel(const _Float16* __restrict__ Ctx,
                     const float* __restrict__ Wo, float* __restrict__ Out)
{
    __shared__ _Float16 Ash[128][40];
    __shared__ _Float16 Bsh[64][40];

    const int tid     = threadIdx.x;
    const int lane    = tid & 31;
    const int wave    = tid >> 5;
    const int half_id = lane >> 4;
    const int lrow    = lane & 15;

    const int n0 = blockIdx.x * 64;
    const int m0 = blockIdx.y * 128;

    v8f acc[4] = {};

    const int ar = tid >> 1, ac = (tid & 1) * 16;  // A staging (f16 copy)
    const int br = tid >> 2, bc = (tid & 3) * 8;   // B staging (f32 -> f16)

    for (int k0 = 0; k0 < Dd; k0 += 32) {
        __syncthreads();
        {   // A: 128 x 32 halves
            const _Float16* cp = Ctx + (size_t)(m0 + ar) * Dd + k0 + ac;
            __builtin_prefetch(cp + 32, 0, 1);
            *(uint4*)&Ash[ar][ac]     = *(const uint4*)(cp);
            *(uint4*)&Ash[ar][ac + 8] = *(const uint4*)(cp + 8);
        }
        {   // B: 64 x 32 (f32 -> f16)
            const float* wp = Wo + (size_t)(n0 + br) * Dd + k0 + bc;
            float4 w0 = *(const float4*)(wp);
            float4 w1 = *(const float4*)(wp + 4);
            __builtin_prefetch(wp + 32, 0, 1);
            Bsh[br][bc + 0] = (_Float16)w0.x;
            Bsh[br][bc + 1] = (_Float16)w0.y;
            Bsh[br][bc + 2] = (_Float16)w0.z;
            Bsh[br][bc + 3] = (_Float16)w0.w;
            Bsh[br][bc + 4] = (_Float16)w1.x;
            Bsh[br][bc + 5] = (_Float16)w1.y;
            Bsh[br][bc + 6] = (_Float16)w1.z;
            Bsh[br][bc + 7] = (_Float16)w1.w;
        }
        __syncthreads();

        v16h a = load_fragA(&Ash[wave * 16][0], 40, lrow, half_id, 0);
#pragma unroll
        for (int nt = 0; nt < 4; ++nt) {
            v16h b = load_fragB(&Bsh[nt * 16][0], 40, lrow, half_id, 0);
            acc[nt] = wmma16(a, b, acc[nt]);
        }
    }

#pragma unroll
    for (int nt = 0; nt < 4; ++nt) {
#pragma unroll
        for (int r = 0; r < 8; ++r) {
            int m = m0 + wave * 16 + r + 8 * half_id;
            int n = n0 + nt * 16 + lrow;
            Out[(size_t)m * Dd + n] = acc[nt][r];
        }
    }
}

} // namespace ecsa

// ---------------------------------------------------------------------------
// Launcher
// ---------------------------------------------------------------------------
extern "C" void kernel_launch(void* const* d_in, const int* in_sizes, int n_in,
                              void* d_out, int out_size, void* d_ws, size_t ws_size,
                              hipStream_t stream) {
    (void)in_sizes; (void)n_in; (void)out_size; (void)ws_size;
    using namespace ecsa;

    const float* X    = (const float*)d_in[0];   // (B,S,D)
    const float* mask = (const float*)d_in[1];   // (S,S)
    const float* Wq   = (const float*)d_in[2];   // (H*DK, D)
    const float* Wk   = (const float*)d_in[3];
    const float* Wv   = (const float*)d_in[4];
    const float* Wo   = (const float*)d_in[5];   // (D, H*DK)
    float* Out = (float*)d_out;                  // (B,S,D)

    const size_t nQKV = (size_t)Bb * Hh * Ss * DKk;   // 8388608 halves
    _Float16* Qb = (_Float16*)d_ws;
    _Float16* Kb = Qb + nQKV;
    _Float16* Vb = Kb + nQKV;   // stored transposed (B,H,DK,S)
    _Float16* Cb = Vb + nQKV;   // ctx (B,S,H*DK)

    dim3 blk(256);
    dim3 gemm_grid(Dd / 64, (Bb * Ss) / 128);    // (16, 64)

    // QKV projections (Q pre-scaled by 1/sqrt(DK) = 0.125)
    qkv_proj_kernel<<<gemm_grid, blk, 0, stream>>>(X, Wq, Qb, 0.125f, DKk, 1);
    qkv_proj_kernel<<<gemm_grid, blk, 0, stream>>>(X, Wk, Kb, 1.0f,  DKk, 1);
    qkv_proj_kernel<<<gemm_grid, blk, 0, stream>>>(X, Wv, Vb, 1.0f,  1, Ss);

    // Flash attention: grid (B*H, S/128)
    attn_kernel<<<dim3(Bb * Hh, Ss / 128), blk, 0, stream>>>(Qb, Kb, Vb, mask, Cb);

    // Output projection
    out_proj_kernel<<<gemm_grid, blk, 0, stream>>>(Cb, Wo, Out);
}
